// DARNN_86122684220300
// MI455X (gfx1250) — compile-verified, matrix operand order checked by reading, not run on hardware
//
#include <hip/hip_runtime.h>
#include <hip/hip_bf16.h>
#include <math.h>

#define B_ 4096
#define T_ 128
#define N_ 64
#define M_ 128
#define P_ 128
#define G_ 512   // 4*M gates

typedef __attribute__((ext_vector_type(16))) __bf16 v16bf;
typedef __attribute__((ext_vector_type(8)))  __bf16 v8bf;
typedef __attribute__((ext_vector_type(8)))  float  v8f;

#define DEV __device__ __forceinline__

// ---------------- CDNA5 helpers (gfx1250, wave32) ----------------

DEV int lane_id() { return (int)(threadIdx.x & 31u); }

// native gfx1250 transcendental tanh
DEV float tanh_fast(float x) {
  float r;
  asm("v_tanh_f32 %0, %1" : "=v"(r) : "v"(x));
  return r;
}

DEV float sigm(float x) { return 1.0f / (1.0f + __expf(-x)); }

// D = A(16x32 bf16) * B(32x16 bf16) + C, f32 accumulate
DEV v8f wmma_bf16(v16bf a, v16bf b, v8f c) {
  return __builtin_amdgcn_wmma_f32_16x16x32_bf16(
      /*neg_a=*/false, a, /*neg_b=*/false, b,
      /*c_mod=*/(short)0, c, /*reuse_a=*/false, /*reuse_b=*/false);
}

// A fragment: 16x32 tile, row-major bf16 source, row stride ld (elements).
// lane 0-15: M=lane, K=0..7,16..23 ; lane 16-31: M=lane-16, K=8..15,24..31
DEV v16bf frag_a_lds(const __bf16* src, int ld) {
  const int lane = lane_id();
  const int row  = lane & 15;
  const int kb   = (lane >> 4) << 3;
  const __bf16* p = src + row * ld + kb;
  v8bf lo = *(const v8bf*)(p);        // k = kb .. kb+7
  v8bf hi = *(const v8bf*)(p + 16);   // k = kb+16 .. kb+23
  return __builtin_shufflevector(lo, hi, 0, 1, 2, 3, 4, 5, 6, 7,
                                         8, 9, 10, 11, 12, 13, 14, 15);
}

// A fragment from a transposed bf16 source: A[m][k] = src[k*ld + m] (gather)
DEV v16bf frag_a_tr_lds(const __bf16* src, int ld) {
  const int lane = lane_id();
  const int row  = lane & 15;
  const int kb   = (lane >> 4) << 3;
  v16bf a;
#pragma unroll
  for (int j = 0; j < 16; ++j) {
    const int k = kb + (j & 7) + ((j >> 3) << 4);
    a[j] = src[k * ld + row];
  }
  return a;
}

// A fragment from row-major global f32 (vectorized float4 loads + convert)
DEV v16bf frag_a_gf32(const float* src, int ld) {
  const int lane = lane_id();
  const int row  = lane & 15;
  const int kb   = (lane >> 4) << 3;
  const float* p = src + (size_t)row * ld + kb;
  const float4 a0 = ((const float4*)p)[0];
  const float4 a1 = ((const float4*)p)[1];
  const float4 b0 = ((const float4*)(p + 16))[0];
  const float4 b1 = ((const float4*)(p + 16))[1];
  v16bf a;
  a[0] = (__bf16)a0.x;  a[1] = (__bf16)a0.y;  a[2] = (__bf16)a0.z;  a[3] = (__bf16)a0.w;
  a[4] = (__bf16)a1.x;  a[5] = (__bf16)a1.y;  a[6] = (__bf16)a1.z;  a[7] = (__bf16)a1.w;
  a[8] = (__bf16)b0.x;  a[9] = (__bf16)b0.y;  a[10] = (__bf16)b0.z; a[11] = (__bf16)b0.w;
  a[12] = (__bf16)b1.x; a[13] = (__bf16)b1.y; a[14] = (__bf16)b1.z; a[15] = (__bf16)b1.w;
  return a;
}

// B fragment from pre-swizzled fragment-major storage (global or LDS):
// fragment f occupies 512 bf16; lane's 16 values contiguous -> one 32B load.
DEV v16bf frag_b_sw(const __bf16* base, int frag) {
  return *(const v16bf*)(base + (size_t)frag * 512 + lane_id() * 16);
}

// ======================================================================
// Weight pre-swizzle: W (K x Nw, f32 row-major) -> fragment-major bf16.
// ======================================================================
__global__ void k_swz(const float* __restrict__ W, int Nw, int nNT, int nfrag,
                      __bf16* __restrict__ out) {
  const int wave = (int)(threadIdx.x >> 5);
  const int f = blockIdx.x * 4 + wave;
  if (f >= nfrag) return;
  const int lane = lane_id();
  const int kk = f / nNT, nt = f % nNT;
  const int col = nt * 16 + (lane & 15);
  const int kb  = kk * 32 + ((lane >> 4) << 3);
  v16bf b;
#pragma unroll
  for (int j = 0; j < 16; ++j) {
    const int k = kb + (j & 7) + ((j >> 3) << 4);
    b[j] = (__bf16)W[(size_t)k * Nw + col];
  }
  *(v16bf*)(out + (size_t)f * 512 + lane * 16) = b;
}

// ======================================================================
// Kernel 1: ux[b][n][t] = sum_s X[b][s][n] * Ue[s][t] + bue[t]
// ======================================================================
__global__ void k_ux(const float* __restrict__ X, const __bf16* __restrict__ Ues,
                     const float* __restrict__ bue, float* __restrict__ ux) {
  __shared__ __bf16 Xb[T_ * N_];               // X[b] as bf16, [s][n], 16KB
  const int b = blockIdx.x;
  const float* Xsrc = X + (size_t)b * T_ * N_;
  for (int i = threadIdx.x; i < T_ * N_; i += blockDim.x) Xb[i] = (__bf16)Xsrc[i];
  __syncthreads();

  const int wave = (int)(threadIdx.x >> 5);
  const int n0   = wave * 16;
  const int lane = lane_id();
  const int col  = lane & 15;
  const int mb   = (lane >> 4) << 3;

  for (int nt = 0; nt < 8; ++nt) {             // t tiles (T=128)
    const int t0 = nt * 16;
    v8f acc = {};
#pragma unroll
    for (int kk = 0; kk < 4; ++kk) {           // K = s = 128
      const int k0 = kk * 32;
      v16bf a  = frag_a_tr_lds(Xb + (size_t)k0 * N_ + n0, N_); // A[n][s]=Xb[s][n]
      v16bf bb = frag_b_sw(Ues, kk * 8 + nt);
      acc = wmma_bf16(a, bb, acc);
    }
    const float bias = bue[t0 + col];
    float* dst = ux + ((size_t)b * N_ + n0) * T_ + t0;
#pragma unroll
    for (int r = 0; r < 8; ++r) dst[(size_t)(mb + r) * T_ + col] = acc[r] + bias;
  }
}

// ======================================================================
// Kernel 2: persistent encoder. 128 threads = 4 waves, wave owns 16 batch rows.
// LDS = Whe (swizzled, shared) + ve + per-wave state.
// ======================================================================
#define ENC_WHE_BYTES  131072          // 128 frags * 1KB
#define ENC_VE_BYTES   512
#define ENC_WAVE_BYTES 34816
__global__ void k_encoder(const float* __restrict__ X,
                          const __bf16* __restrict__ We1s, const float* __restrict__ be1,
                          const float* __restrict__ ve,  const float* __restrict__ bve,
                          const __bf16* __restrict__ Wxes, const __bf16* __restrict__ Whes,
                          const float* __restrict__ bbe,
                          const float* __restrict__ ux,  float* __restrict__ Xe) {
  extern __shared__ char smem[];
  const int wave  = (int)(threadIdx.x >> 5);
  const int lane  = lane_id();
  __bf16* WheL = (__bf16*)smem;                    // 128 swizzled frags, 128KB
  float*  veL  = (float*)(smem + ENC_WHE_BYTES);   // 128 floats
  char* wbase = smem + ENC_WHE_BYTES + ENC_VE_BYTES + (size_t)wave * ENC_WAVE_BYTES;
  __bf16* hc   = (__bf16*)(wbase);                 // [16][256] bf16: h | c
  __bf16* hnew = (__bf16*)(wbase + 8192);          // [16][128]
  __bf16* xt   = (__bf16*)(wbase + 12288);         // [16][64]
  float*  cst  = (float*)(wbase + 14336);          // [16][128] cell state f32
  float*  hs   = (float*)(wbase + 22528);          // [16][128]
  float*  alp  = (float*)(wbase + 30720);          // [16][64]

  // stage Whe (swizzled) + ve into LDS
  {
    const v16bf* s = (const v16bf*)Whes;
    v16bf* d = (v16bf*)WheL;
    for (int i = threadIdx.x; i < ENC_WHE_BYTES / 32; i += blockDim.x) d[i] = s[i];
    if (threadIdx.x < 128) veL[threadIdx.x] = ve[threadIdx.x];
  }

  const int brow0 = (blockIdx.x * 4 + wave) * 16;
  const int col   = lane & 15;
  const int mb    = (lane >> 4) << 3;
  const float bve0 = bve[0];

  for (int i = lane; i < 16 * 256; i += 32) hc[i] = (__bf16)0.0f;
  for (int i = lane; i < 16 * 128; i += 32) cst[i] = 0.0f;
  __syncthreads();

  for (int t = 0; t < T_; ++t) {
    // --- 1) hs = [h,c] @ We1 + be1   (16x256 @ 256x128) ---
    for (int nt = 0; nt < 8; ++nt) {
      v8f acc = {};
#pragma unroll
      for (int kk = 0; kk < 8; ++kk) {
        v16bf a  = frag_a_lds(hc + kk * 32, 256);
        v16bf bb = frag_b_sw(We1s, kk * 8 + nt);
        acc = wmma_bf16(a, bb, acc);
      }
      const float bias = be1[nt * 16 + col];
#pragma unroll
      for (int r = 0; r < 8; ++r) hs[(mb + r) * T_ + nt * 16 + col] = acc[r] + bias;
    }

    // --- 2) input attention: e = tanh(hs + ux) @ ve ; softmax over n ---
    for (int p = lane; p < 16 * 64; p += 32) {
      const int row = p >> 6, n = p & 63;
      const float4* uxp = (const float4*)(ux + ((size_t)(brow0 + row) * N_ + n) * T_);
      const float4* hsr = (const float4*)(hs + row * T_);
      const float4* vv  = (const float4*)veL;
      float e = bve0;
      for (int q = 0; q < T_ / 4; ++q) {
        const float4 u = uxp[q], h4 = hsr[q], v4 = vv[q];
        e += tanh_fast(h4.x + u.x) * v4.x;
        e += tanh_fast(h4.y + u.y) * v4.y;
        e += tanh_fast(h4.z + u.z) * v4.z;
        e += tanh_fast(h4.w + u.w) * v4.w;
      }
      alp[row * 64 + n] = e;
    }
    if (lane < 16) {
      float* ar = alp + lane * 64;
      float mx = ar[0];
      for (int n = 1; n < 64; ++n) mx = fmaxf(mx, ar[n]);
      float s = 0.0f;
      for (int n = 0; n < 64; ++n) { float v = __expf(ar[n] - mx); ar[n] = v; s += v; }
      const float inv = 1.0f / s;
      for (int n = 0; n < 64; ++n) ar[n] *= inv;
    }
    for (int p = lane; p < 16 * 64; p += 32) {
      const int row = p >> 6, n = p & 63;
      const float xv = X[((size_t)(brow0 + row) * T_ + t) * N_ + n];
      xt[row * 64 + n] = (__bf16)(alp[row * 64 + n] * xv);
    }

    // --- 3) LSTM gates: z = x_tilde@Wxe + h@Whe + bbe  (out 16x512) ---
    for (int j = 0; j < 8; ++j) {              // tile over M (gate columns)
      v8f acc[4];
#pragma unroll
      for (int g = 0; g < 4; ++g) {
        const int gc = g * 128 + j * 16 + col;
        const float bv = bbe[gc];
#pragma unroll
        for (int r = 0; r < 8; ++r) acc[g][r] = bv;
      }
#pragma unroll
      for (int kk = 0; kk < 2; ++kk) {         // K = 64 (x_tilde)
        v16bf a = frag_a_lds(xt + kk * 32, 64);
#pragma unroll
        for (int g = 0; g < 4; ++g) {
          v16bf bb = frag_b_sw(Wxes, kk * 32 + g * 8 + j);
          acc[g] = wmma_bf16(a, bb, acc[g]);
        }
      }
#pragma unroll
      for (int kk = 0; kk < 4; ++kk) {         // K = 128 (h)
        v16bf a = frag_a_lds(hc + kk * 32, 256);
#pragma unroll
        for (int g = 0; g < 4; ++g) {
          v16bf bb = frag_b_sw(WheL, kk * 32 + g * 8 + j);   // from LDS
          acc[g] = wmma_bf16(a, bb, acc[g]);
        }
      }
      const int m = j * 16 + col;
#pragma unroll
      for (int r = 0; r < 8; ++r) {
        const int row = mb + r;
        const float iv = sigm(acc[0][r]);
        const float fv = sigm(acc[1][r]);
        const float gv = tanh_fast(acc[2][r]);
        const float ov = sigm(acc[3][r]);
        const float cn = fv * cst[row * 128 + m] + iv * gv;
        const float hn = ov * tanh_fast(cn);
        cst[row * 128 + m]      = cn;
        hc[row * 256 + 128 + m] = (__bf16)cn;   // c region (not read this stage)
        hnew[row * 128 + m]     = (__bf16)hn;   // defer h commit
        Xe[((size_t)(brow0 + row) * T_ + t) * M_ + m] = hn;
      }
    }
    // commit h
    for (int i = lane; i < 16 * 128; i += 32) {
      const int row = i >> 7, m = i & 127;
      hc[row * 256 + m] = hnew[i];
    }
    __syncthreads();
  }
}

// ======================================================================
// Kernel 3: uh = X_enc @ Ud + bud over B*T rows.
// ======================================================================
__global__ void k_uh(const float* __restrict__ Xe, const __bf16* __restrict__ Uds,
                     const float* __restrict__ bud, float* __restrict__ uh) {
  const int wave    = (int)(threadIdx.x >> 5);
  const size_t row0 = ((size_t)blockIdx.x * 4 + wave) * 16;
  const int lane = lane_id();
  const int col  = lane & 15;
  const int mb   = (lane >> 4) << 3;
  for (int nt = 0; nt < 8; ++nt) {
    v8f acc = {};
#pragma unroll
    for (int kk = 0; kk < 4; ++kk) {
      v16bf a  = frag_a_gf32(Xe + row0 * M_ + kk * 32, M_);
      v16bf bb = frag_b_sw(Uds, kk * 8 + nt);
      acc = wmma_bf16(a, bb, acc);
    }
    const float bias = bud[nt * 16 + col];
    float* dst = uh + row0 * M_ + nt * 16;
#pragma unroll
    for (int r = 0; r < 8; ++r) dst[(size_t)(mb + r) * M_ + col] = acc[r] + bias;
  }
}

// ======================================================================
// Kernel 4: persistent decoder + output head.
// LDS = Whd (swizzled, shared) + vd + per-wave state.
// ======================================================================
#define DEC_WHD_BYTES  131072
#define DEC_VD_BYTES   512
#define DEC_WAVE_BYTES 45312
__global__ void k_decoder(const float* __restrict__ Y,
                          const __bf16* __restrict__ Wds, const float* __restrict__ bd,
                          const float* __restrict__ vd,  const float* __restrict__ bvd,
                          const float* __restrict__ Wy,  const float* __restrict__ by,
                          const float* __restrict__ Wxd, const __bf16* __restrict__ Whds,
                          const float* __restrict__ bbd,
                          const float* __restrict__ Wo1, const float* __restrict__ bo1,
                          const float* __restrict__ Wo2, const float* __restrict__ bo2,
                          const float* __restrict__ Xe,  const float* __restrict__ uh,
                          float* __restrict__ out) {
  extern __shared__ char smem[];
  const int wave = (int)(threadIdx.x >> 5);
  const int lane = lane_id();
  __bf16* WhdL = (__bf16*)smem;                    // 128 swizzled frags, 128KB
  float*  vdL  = (float*)(smem + DEC_WHD_BYTES);   // 128 floats
  char* wbase = smem + DEC_WHD_BYTES + DEC_VD_BYTES + (size_t)wave * DEC_WAVE_BYTES;
  __bf16* hc   = (__bf16*)(wbase);                 // [16][256] h | c (bf16)
  __bf16* hnew = (__bf16*)(wbase + 8192);          // [16][128] (also last h for head)
  float*  cst  = (float*)(wbase + 12288);          // [16][128] cell f32
  float*  dsb  = (float*)(wbase + 20480);          // [16][128]
  float*  beta = (float*)(wbase + 28672);          // [16][128]
  float*  ct   = (float*)(wbase + 36864);          // [16][128] context
  float*  ytl  = (float*)(wbase + 45056);          // [16]

  {
    const v16bf* s = (const v16bf*)Whds;
    v16bf* d = (v16bf*)WhdL;
    for (int i = threadIdx.x; i < DEC_WHD_BYTES / 32; i += blockDim.x) d[i] = s[i];
    if (threadIdx.x < 128) vdL[threadIdx.x] = vd[threadIdx.x];
  }

  const int brow0 = (blockIdx.x * 4 + wave) * 16;
  const int col   = lane & 15;
  const int mb    = (lane >> 4) << 3;
  const float bvd0 = bvd[0];

  for (int i = lane; i < 16 * 256; i += 32) hc[i] = (__bf16)0.0f;
  for (int i = lane; i < 16 * 128; i += 32) { cst[i] = 0.0f; ct[i] = 0.0f; hnew[i] = (__bf16)0.0f; }
  __syncthreads();

  for (int t = 0; t < T_ - 1; ++t) {
    // --- 1) ds = [h,c] @ Wd + bd  (16x256 @ 256x128) ---
    for (int nt = 0; nt < 8; ++nt) {
      v8f acc = {};
#pragma unroll
      for (int kk = 0; kk < 8; ++kk) {
        v16bf a  = frag_a_lds(hc + kk * 32, 256);
        v16bf bb = frag_b_sw(Wds, kk * 8 + nt);
        acc = wmma_bf16(a, bb, acc);
      }
      const float bias = bd[nt * 16 + col];
#pragma unroll
      for (int r = 0; r < 8; ++r) dsb[(mb + r) * M_ + nt * 16 + col] = acc[r] + bias;
    }

    // --- 2) temporal attention: e = tanh(ds + uh) @ vd ; softmax over T ---
    for (int p = lane; p < 16 * 128; p += 32) {
      const int row = p >> 7, tt = p & 127;
      const float4* uhp = (const float4*)(uh + ((size_t)(brow0 + row) * T_ + tt) * M_);
      const float4* dr  = (const float4*)(dsb + row * M_);
      const float4* vv  = (const float4*)vdL;
      float e = bvd0;
      for (int q = 0; q < M_ / 4; ++q) {
        const float4 u = uhp[q], d4 = dr[q], v4 = vv[q];
        e += tanh_fast(d4.x + u.x) * v4.x;
        e += tanh_fast(d4.y + u.y) * v4.y;
        e += tanh_fast(d4.z + u.z) * v4.z;
        e += tanh_fast(d4.w + u.w) * v4.w;
      }
      beta[p] = e;
    }
    if (lane < 16) {
      float* br = beta + lane * 128;
      float mx = br[0];
      for (int n = 1; n < 128; ++n) mx = fmaxf(mx, br[n]);
      float s = 0.0f;
      for (int n = 0; n < 128; ++n) { float v = __expf(br[n] - mx); br[n] = v; s += v; }
      const float inv = 1.0f / s;
      for (int n = 0; n < 128; ++n) br[n] *= inv;
    }
    // --- 3) context c_t = beta @ X_enc[b]  (per-row matvec, float4 along m) ---
    for (int p = lane; p < 16 * 32; p += 32) {
      const int row = p >> 5, mg = p & 31;
      const float4* xep = (const float4*)(Xe + (size_t)(brow0 + row) * T_ * M_) + mg;
      const float* br   = beta + row * 128;
      float4 s; s.x = 0.f; s.y = 0.f; s.z = 0.f; s.w = 0.f;
      for (int tt = 0; tt < T_; ++tt) {
        const float bv = br[tt];
        const float4 x = xep[(size_t)tt * (M_ / 4)];
        s.x += bv * x.x; s.y += bv * x.y; s.z += bv * x.z; s.w += bv * x.w;
      }
      ((float4*)ct)[row * 32 + mg] = s;
    }
    // --- 4) y_tilde = [y_t, c_t] @ Wy + by ---
    if (lane < 16) {
      const float yv = Y[(size_t)(brow0 + lane) * T_ + t];
      float s = yv * Wy[0] + by[0];
      const float* cr = ct + lane * 128;
      for (int m = 0; m < M_; ++m) s += cr[m] * Wy[1 + m];
      ytl[lane] = s;
    }
    // --- 5) LSTM gates: z = y_tilde*Wxd + h@Whd + bbd ---
    for (int j = 0; j < 8; ++j) {
      v8f acc[4];
#pragma unroll
      for (int g = 0; g < 4; ++g) {
        const int gc = g * 128 + j * 16 + col;
        const float bv = bbd[gc];
        const float wx = Wxd[gc];
#pragma unroll
        for (int r = 0; r < 8; ++r) acc[g][r] = bv + ytl[mb + r] * wx;
      }
#pragma unroll
      for (int kk = 0; kk < 4; ++kk) {
        v16bf a = frag_a_lds(hc + kk * 32, 256);
#pragma unroll
        for (int g = 0; g < 4; ++g) {
          v16bf bb = frag_b_sw(WhdL, kk * 32 + g * 8 + j);   // from LDS
          acc[g] = wmma_bf16(a, bb, acc[g]);
        }
      }
      const int m = j * 16 + col;
#pragma unroll
      for (int r = 0; r < 8; ++r) {
        const int row = mb + r;
        const float iv = sigm(acc[0][r]);
        const float fv = sigm(acc[1][r]);
        const float gv = tanh_fast(acc[2][r]);
        const float ov = sigm(acc[3][r]);
        const float cn = fv * cst[row * 128 + m] + iv * gv;
        const float hn = ov * tanh_fast(cn);
        cst[row * 128 + m]      = cn;
        hc[row * 256 + 128 + m] = (__bf16)cn;
        hnew[row * 128 + m]     = (__bf16)hn;
      }
    }
    for (int i = lane; i < 16 * 128; i += 32) {
      const int row = i >> 7, m = i & 127;
      hc[row * 256 + m] = hnew[i];
    }
    __syncthreads();
  }

  // --- output head: y_hat = ([h, c_t] @ Wo1 + bo1) @ Wo2 + bo2 ---
  if (lane < 16) {
    const int row = lane;
    float y = bo2[0];
    for (int p = 0; p < P_; ++p) {
      float s = bo1[p];
      const float* w1 = Wo1 + p;               // Wo1[(P+M) x P], k-major
      const __bf16* hr = hnew + row * 128;     // last hidden (bf16)
      const float*  cr = ct + row * 128;
      for (int k = 0; k < 128; ++k) s += (float)hr[k] * w1[(size_t)k * P_];
      for (int k = 0; k < 128; ++k) s += cr[k] * w1[(size_t)(128 + k) * P_];
      y += s * Wo2[p];
    }
    out[brow0 + row] = y;
  }
}

// ======================================================================
extern "C" void kernel_launch(void* const* d_in, const int* in_sizes, int n_in,
                              void* d_out, int out_size, void* d_ws, size_t ws_size,
                              hipStream_t stream) {
  (void)in_sizes; (void)n_in; (void)out_size; (void)ws_size;
  const float* X   = (const float*)d_in[0];
  const float* Y   = (const float*)d_in[1];
  const float* We1 = (const float*)d_in[2];
  const float* be1 = (const float*)d_in[3];
  const float* Ue  = (const float*)d_in[4];
  const float* bue = (const float*)d_in[5];
  const float* ve  = (const float*)d_in[6];
  const float* bve = (const float*)d_in[7];
  const float* Wxe = (const float*)d_in[8];
  const float* Whe = (const float*)d_in[9];
  const float* bbe = (const float*)d_in[10];
  const float* Wd  = (const float*)d_in[11];
  const float* bd  = (const float*)d_in[12];
  const float* Ud  = (const float*)d_in[13];
  const float* bud = (const float*)d_in[14];
  const float* vd  = (const float*)d_in[15];
  const float* bvd = (const float*)d_in[16];
  const float* Wy  = (const float*)d_in[17];
  const float* by  = (const float*)d_in[18];
  const float* Wxd = (const float*)d_in[19];
  const float* Whd = (const float*)d_in[20];
  const float* bbd = (const float*)d_in[21];
  const float* Wo1 = (const float*)d_in[22];
  const float* bo1 = (const float*)d_in[23];
  const float* Wo2 = (const float*)d_in[24];
  const float* bo2 = (const float*)d_in[25];

  float* ws  = (float*)d_ws;
  float* ux  = ws;                                   // B*N*T
  float* Xe  = ux + (size_t)B_ * N_ * T_;            // B*T*M
  float* uh  = Xe + (size_t)B_ * T_ * M_;            // B*T*M
  float* out = (float*)d_out;

  // swizzled bf16 weights (fragment-major) in workspace tail
  __bf16* sw = (__bf16*)(uh + (size_t)B_ * T_ * M_);
  __bf16* Ues  = sw;                 // 32 frags
  __bf16* We1s = Ues  + 32 * 512;    // 64 frags
  __bf16* Wxes = We1s + 64 * 512;    // 64 frags
  __bf16* Whes = Wxes + 64 * 512;    // 128 frags
  __bf16* Uds  = Whes + 128 * 512;   // 32 frags
  __bf16* Wds  = Uds  + 32 * 512;    // 64 frags
  __bf16* Whds = Wds  + 64 * 512;    // 128 frags

  // one-time weight swizzle (fragment-major bf16)
  k_swz<<<8, 128, 0, stream>>>(Ue, 128, 8, 32, Ues);
  k_swz<<<16, 128, 0, stream>>>(We1, 128, 8, 64, We1s);
  k_swz<<<16, 128, 0, stream>>>(Wxe, 512, 32, 64, Wxes);
  k_swz<<<32, 128, 0, stream>>>(Whe, 512, 32, 128, Whes);
  k_swz<<<8, 128, 0, stream>>>(Ud, 128, 8, 32, Uds);
  k_swz<<<16, 128, 0, stream>>>(Wd, 128, 8, 64, Wds);
  k_swz<<<32, 128, 0, stream>>>(Whd, 512, 32, 128, Whds);

  k_ux<<<B_, 128, 0, stream>>>(X, Ues, bue, ux);
  k_encoder<<<B_ / 64, 128, ENC_WHE_BYTES + ENC_VE_BYTES + 4 * ENC_WAVE_BYTES, stream>>>(
      X, We1s, be1, ve, bve, Wxes, Whes, bbe, ux, Xe);
  k_uh<<<(B_ * T_) / 64, 128, 0, stream>>>(Xe, Uds, bud, uh);
  k_decoder<<<B_ / 64, 128, DEC_WHD_BYTES + DEC_VD_BYTES + 4 * DEC_WAVE_BYTES, stream>>>(
      Y, Wds, bd, vd, bvd, Wy, by, Wxd, Whds, bbd, Wo1, bo1, Wo2, bo2, Xe, uh, out);
}
